// GPTModel_60455959659087
// MI455X (gfx1250) — matrix operand order repaired and checked by reading, not compiled, not run on hardware
//
#include <hip/hip_runtime.h>
#include <hip/hip_bf16.h>
#include <cstdint>
#include <cstddef>

// ---------------------------------------------------------------------------
// GPT forward for MI455X (gfx1250, wave32, WMMA).
// bf16 WMMA (v_wmma_f32_16x16x32_bf16) with fp32 accumulation, fp32 residual
// stream. Weights converted fp32->bf16 while staging to LDS. bf16 tiles are
// staged with GLOBAL_LOAD_ASYNC_TO_LDS_B128 (ASYNCcnt) when available.
// ---------------------------------------------------------------------------

typedef __bf16 bf16_t;
typedef __attribute__((ext_vector_type(16))) __bf16 v16bf;
typedef __attribute__((ext_vector_type(8)))  __bf16 v8bf;
typedef __attribute__((ext_vector_type(4)))  __bf16 v4bf;
typedef __attribute__((ext_vector_type(8)))  float  v8f;

#define B_   2
#define T_   1024
#define D_   1024
#define H_   16
#define HD_  64
#define L_   4
#define V_   32000
#define F_   4096
#define M_   (B_ * T_)   // 2048 token rows

// --- async global->LDS copy (CDNA5), with safe fallback --------------------
#ifndef USE_ASYNC_LDS
#if defined(__has_builtin)
#if __has_builtin(__builtin_amdgcn_global_load_async_to_lds_b128) && \
    __has_builtin(__builtin_amdgcn_s_wait_asynccnt)
#define USE_ASYNC_LDS 1
#endif
#endif
#endif
#ifndef USE_ASYNC_LDS
#define USE_ASYNC_LDS 0
#endif

#if USE_ASYNC_LDS
// Builtin parameter type (from hipcc diagnostic): pointer to GCC-style
// 16-byte int vector, global (AS1) source and LDS (AS3) destination.
typedef int v4i_ __attribute__((vector_size(4 * sizeof(int))));
static __device__ inline void async_cp16(const void* g, void* l) {
    __builtin_amdgcn_global_load_async_to_lds_b128(
        (__attribute__((address_space(1))) v4i_*)g,
        (__attribute__((address_space(3))) v4i_*)l, 0, 0);
}
static __device__ inline void async_wait0() {
    __builtin_amdgcn_s_wait_asynccnt(0);
}
#else
static __device__ inline void async_cp16(const void* g, void* l) {
    *(v8bf*)l = *(const v8bf*)g;
}
static __device__ inline void async_wait0() {}
#endif

// fp32 -> bf16 round-to-nearest-even (pure bit math)
static __device__ inline bf16_t f2bf(float f) {
    unsigned u = __float_as_uint(f);
    unsigned r = (u + 0x7FFFu + ((u >> 16) & 1u)) >> 16;
    unsigned short s = (unsigned short)r;
    return __builtin_bit_cast(bf16_t, s);
}

static __device__ inline v16bf cat16(v8bf lo, v8bf hi) {
    return __builtin_shufflevector(lo, hi, 0,1,2,3,4,5,6,7,8,9,10,11,12,13,14,15);
}

// A-operand fragment (16x32 bf16, ISA 7.12.2): lane(half,m) holds
// K = half*8 + [0..7] and K = 16 + half*8 + [0..7]  -> two b128 LDS loads.
static __device__ inline v16bf ld_frag_a(const bf16_t* rowp, int hf) {
    v8bf lo = *(const v8bf*)(rowp + hf * 8);
    v8bf hi = *(const v8bf*)(rowp + 16 + hf * 8);
    return cat16(lo, hi);
}

// B-operand fragment (32x16 bf16): lane(half,n) holds K = half*8 + [0..7]
// and K = 16 + half*8 + [0..7]; LDS tile stored [n][k] -> two b128 loads.
static __device__ inline v16bf ld_frag_b(const bf16_t* rowp, int hf) {
    v8bf lo = *(const v8bf*)(rowp + hf * 8);
    v8bf hi = *(const v8bf*)(rowp + 16 + hf * 8);
    return cat16(lo, hi);
}

static __device__ inline v8f wmma_bf16(v16bf a, v16bf b, v8f c) {
    // (neg_a, A, neg_b, B, c_mod, C, reuse_a, reuse_b)
    return __builtin_amdgcn_wmma_f32_16x16x32_bf16(false, a, false, b,
                                                   (short)0, c, false, false);
}

static __device__ inline float gelu_exact(float x) {
    return 0.5f * x * (1.0f + erff(x * 0.70710678118654752f));
}

// ---------------------------------------------------------------------------
// Embedding gather + positional encoding -> fp32 residual stream
// ---------------------------------------------------------------------------
__global__ __launch_bounds__(256)
void embed_kernel(const int* __restrict__ x, const float* __restrict__ emb,
                  const float* __restrict__ pe, float* __restrict__ h) {
    const int row = blockIdx.x;          // b*T + t
    const int t   = row & (T_ - 1);
    const int tok = x[row];
    const int c   = threadIdx.x * 4;     // 256 threads * 4 = 1024 = D
    float4 e = *(const float4*)&emb[(size_t)tok * D_ + c];
    float4 p = *(const float4*)&pe[(size_t)t * D_ + c];
    float4 o = make_float4(e.x + p.x, e.y + p.y, e.z + p.z, e.w + p.w);
    *(float4*)&h[(size_t)row * D_ + c] = o;
}

// ---------------------------------------------------------------------------
// LayerNorm: fp32 in -> bf16 out (GEMM A-operand)
// ---------------------------------------------------------------------------
__global__ __launch_bounds__(256)
void layernorm_kernel(const float* __restrict__ x, const float* __restrict__ g,
                      const float* __restrict__ bta, bf16_t* __restrict__ out) {
    const int row  = blockIdx.x;
    const int tid  = threadIdx.x;
    const int lane = tid & 31, wv = tid >> 5;
    const int c = tid * 4;
    float4 v = *(const float4*)&x[(size_t)row * D_ + c];
    float s = v.x + v.y + v.z + v.w;
    float q = v.x * v.x + v.y * v.y + v.z * v.z + v.w * v.w;
    #pragma unroll
    for (int off = 16; off >= 1; off >>= 1) {
        s += __shfl_xor(s, off, 32);
        q += __shfl_xor(q, off, 32);
    }
    __shared__ float ss[8], sq[8];
    if (lane == 0) { ss[wv] = s; sq[wv] = q; }
    __syncthreads();
    float S = 0.f, Q = 0.f;
    #pragma unroll
    for (int i = 0; i < 8; ++i) { S += ss[i]; Q += sq[i]; }
    const float mean = S * (1.0f / D_);
    const float var  = Q * (1.0f / D_) - mean * mean;
    const float inv  = rsqrtf(var + 1e-5f);
    float4 gw = *(const float4*)&g[c];
    float4 gb = *(const float4*)&bta[c];
    bf16_t* o = out + (size_t)row * D_ + c;
    o[0] = f2bf((v.x - mean) * inv * gw.x + gb.x);
    o[1] = f2bf((v.y - mean) * inv * gw.y + gb.y);
    o[2] = f2bf((v.z - mean) * inv * gw.z + gb.z);
    o[3] = f2bf((v.w - mean) * inv * gw.w + gb.w);
}

// ---------------------------------------------------------------------------
// Tiled WMMA GEMM: Out[M,N] = epilogue(A[M,K](bf16) @ W[N,K]^T(fp32) + bias)
// MODE 0: outB = v              1: outB = gelu(v)
// MODE 2: outF = resid + v      3: outF = v
// BM=BN=128, BK=64; 256 thr = 8 waves; wave tile 32x64 = 2x4 WMMA tiles,
// 16 WMMAs per barrier pair. A staged via async LDS copy; W converted
// fp32->bf16 through VGPRs.
// ---------------------------------------------------------------------------
#define BM 128
#define BN 128
#define BK 64
#define LDT (BK + 8)   // 72 elems = 144 B rows (16B aligned)

template <int MODE, bool HAS_BIAS>
__global__ __launch_bounds__(256)
void gemm_bf16_kernel(const bf16_t* __restrict__ A, const float* __restrict__ W,
                      const float* __restrict__ bias, int M, int N, int K,
                      const float* __restrict__ resid,
                      bf16_t* __restrict__ outB, float* __restrict__ outF) {
    __shared__ bf16_t As[BM][LDT];   // [m][k]
    __shared__ bf16_t Bs[BN][LDT];   // [n][k]  (W^T staged transposed)

    const int tid  = threadIdx.x;
    const int lane = tid & 31;
    const int hf   = lane >> 4;
    const int ln   = lane & 15;
    const int wid  = tid >> 5;
    const int wm   = wid & 3;        // 4 wave-rows  * 32 rows
    const int wn   = wid >> 2;       // 2 wave-cols  * 64 cols
    const int m0b  = blockIdx.y * BM;
    const int n0b  = blockIdx.x * BN;

    v8f acc[2][4];
    const v8f vz = {0.f,0.f,0.f,0.f,0.f,0.f,0.f,0.f};
    #pragma unroll
    for (int i = 0; i < 2; ++i)
        #pragma unroll
        for (int j = 0; j < 4; ++j) acc[i][j] = vz;

    for (int k0 = 0; k0 < K; k0 += BK) {
        // A tile: 128x64 bf16 -> async copy straight into LDS (no VGPR trip)
        #pragma unroll
        for (int i = 0; i < 4; ++i) {
            int g  = tid + i * 256;
            int r  = g >> 3, c8 = (g & 7) * 8;
            async_cp16(&A[(size_t)(m0b + r) * K + k0 + c8], &As[r][c8]);
        }
        // W tile: 128x64 fp32 -> bf16, float4 loads, 8 per thread
        #pragma unroll
        for (int i = 0; i < 8; ++i) {
            int g  = tid + i * 256;
            int r  = g >> 4, c4 = (g & 15) * 4;
            float4 wv = *(const float4*)&W[(size_t)(n0b + r) * K + k0 + c4];
            v4bf t;
            t[0] = f2bf(wv.x); t[1] = f2bf(wv.y);
            t[2] = f2bf(wv.z); t[3] = f2bf(wv.w);
            *(v4bf*)&Bs[r][c4] = t;
        }
        async_wait0();          // wave-local ASYNCcnt drain before barrier
        __syncthreads();

        #pragma unroll
        for (int st = 0; st < 2; ++st) {
            v16bf a0 = ld_frag_a(&As[wm * 32 + ln][st * 32], hf);
            v16bf a1 = ld_frag_a(&As[wm * 32 + 16 + ln][st * 32], hf);
            #pragma unroll
            for (int nt = 0; nt < 4; ++nt) {
                v16bf b = ld_frag_b(&Bs[wn * 64 + nt * 16 + ln][st * 32], hf);
                acc[0][nt] = wmma_bf16(a0, b, acc[0][nt]);
                acc[1][nt] = wmma_bf16(a1, b, acc[1][nt]);
            }
        }
        __syncthreads();
    }

    // Epilogue: C/D layout -> row = r + hf*8, col = ln  (per 16x16 tile)
    const int m0 = m0b + wm * 32;
    const int n0 = n0b + wn * 64;
    #pragma unroll
    for (int mt = 0; mt < 2; ++mt) {
        #pragma unroll
        for (int nt = 0; nt < 4; ++nt) {
            const int col = n0 + nt * 16 + ln;
            const float bv = HAS_BIAS ? bias[col] : 0.0f;
            #pragma unroll
            for (int r = 0; r < 8; ++r) {
                const int rw = m0 + mt * 16 + r + hf * 8;
                const size_t idx = (size_t)rw * N + col;
                float v = acc[mt][nt][r] + bv;
                if (MODE == 0)      outB[idx] = f2bf(v);
                else if (MODE == 1) outB[idx] = f2bf(gelu_exact(v));
                else if (MODE == 2) outF[idx] = resid[idx] + v;
                else                outF[idx] = v;
            }
        }
    }
}

// ---------------------------------------------------------------------------
// Flash attention: block = (b, h, 64-query tile), 128 thr = 4 waves,
// wave owns 16 query rows. S = Q K^T (WMMA), online causal softmax, O += P V.
// qkv rows: [b*T+t][3*D], q at +0, k at +D, v at +2D, head h at +h*HD.
// ---------------------------------------------------------------------------
__global__ __launch_bounds__(128)
void flash_attn_kernel(const bf16_t* __restrict__ qkv, bf16_t* __restrict__ out) {
    const int qt  = blockIdx.x;          // query tile (64 rows)
    const int bh  = blockIdx.y;
    const int b   = bh / H_, hh = bh % H_;
    const int tid = threadIdx.x;
    const int wv  = tid >> 5;
    const int lane = tid & 31;
    const int hf  = lane >> 4;
    const int ln  = lane & 15;

    __shared__ bf16_t Qs[64][72];        // [query][d]
    __shared__ bf16_t Ks[64][72];        // [key][d]     (B-op for S)
    __shared__ bf16_t Vs[64][72];        // [d][key]     (B-op for O)
    __shared__ bf16_t Ps[4][16][72];     // per-wave P   (A-op for O)

    const size_t rowbase = (size_t)(b * T_) * (3 * D_);

    // Q tile (async copy, drained by first in-loop async_wait0)
    #pragma unroll
    for (int i = 0; i < 4; ++i) {
        int g = tid + i * 128;
        int r = g >> 3, c8 = (g & 7) * 8;
        async_cp16(&qkv[rowbase + (size_t)(qt * 64 + r) * (3 * D_) + hh * HD_ + c8],
                   &Qs[r][c8]);
    }

    float mrow[8], lrow[8];
    v8f of[4];
    const v8f vz = {0.f,0.f,0.f,0.f,0.f,0.f,0.f,0.f};
    #pragma unroll
    for (int r = 0; r < 8; ++r) { mrow[r] = -INFINITY; lrow[r] = 0.f; }
    #pragma unroll
    for (int d = 0; d < 4; ++d) of[d] = vz;

    for (int kt = 0; kt <= qt; ++kt) {
        // K tile: [key][d] straight async copy
        #pragma unroll
        for (int i = 0; i < 4; ++i) {
            int g = tid + i * 128;
            int r = g >> 3, c8 = (g & 7) * 8;
            async_cp16(&qkv[rowbase + (size_t)(kt * 64 + r) * (3 * D_) + D_ + hh * HD_ + c8],
                       &Ks[r][c8]);
        }
        // V tile: transpose to [d][key] (needs VGPR trip)
        #pragma unroll
        for (int i = 0; i < 8; ++i) {
            int g = tid + i * 128;
            int key = g >> 4, c4 = (g & 15) * 4;
            v4bf t = *(const v4bf*)
                &qkv[rowbase + (size_t)(kt * 64 + key) * (3 * D_) + 2 * D_ + hh * HD_ + c4];
            #pragma unroll
            for (int j = 0; j < 4; ++j) Vs[c4 + j][key] = t[j];
        }
        async_wait0();
        __syncthreads();

        // S = Q K^T : M=16 (wave rows), N=64 keys, K=HD=64 (2 steps)
        v8f s[4] = {vz, vz, vz, vz};
        #pragma unroll
        for (int st = 0; st < 2; ++st) {
            v16bf a = ld_frag_a(&Qs[wv * 16 + ln][st * 32], hf);
            #pragma unroll
            for (int nt = 0; nt < 4; ++nt) {
                v16bf kb = ld_frag_b(&Ks[nt * 16 + ln][st * 32], hf);
                s[nt] = wmma_bf16(a, kb, s[nt]);
            }
        }

        // scale + causal mask
        const int qbase = qt * 64 + wv * 16;
        #pragma unroll
        for (int nt = 0; nt < 4; ++nt) {
            const int kj = kt * 64 + nt * 16 + ln;
            #pragma unroll
            for (int r = 0; r < 8; ++r) {
                const int qi = qbase + r + hf * 8;
                float val = s[nt][r] * 0.125f;          // 1/sqrt(64)
                s[nt][r] = (kj > qi) ? -INFINITY : val;
            }
        }

        // online softmax per row r (row lives in the 16-lane half group)
        #pragma unroll
        for (int r = 0; r < 8; ++r) {
            float mx = fmaxf(fmaxf(s[0][r], s[1][r]), fmaxf(s[2][r], s[3][r]));
            #pragma unroll
            for (int off = 1; off < 16; off <<= 1)
                mx = fmaxf(mx, __shfl_xor(mx, off, 32));
            const float mnew = fmaxf(mrow[r], mx);
            const float fac  = __expf(mrow[r] - mnew);
            float rs = 0.f;
            #pragma unroll
            for (int nt = 0; nt < 4; ++nt) {
                float pv = __expf(s[nt][r] - mnew);
                rs += pv;
                Ps[wv][r + hf * 8][nt * 16 + ln] = f2bf(pv);
            }
            #pragma unroll
            for (int off = 1; off < 16; off <<= 1)
                rs += __shfl_xor(rs, off, 32);
            lrow[r] = lrow[r] * fac + rs;
            mrow[r] = mnew;
            #pragma unroll
            for (int d = 0; d < 4; ++d) of[d][r] *= fac;
        }

        // O += P V : M=16, N=HD=64, K=64 keys (2 steps). In-wave DS ordering
        // makes Ps stores visible to the fragment loads below.
        #pragma unroll
        for (int st = 0; st < 2; ++st) {
            v16bf pa = ld_frag_a(&Ps[wv][ln][st * 32], hf);
            #pragma unroll
            for (int dt = 0; dt < 4; ++dt) {
                v16bf vb = ld_frag_b(&Vs[dt * 16 + ln][st * 32], hf);
                of[dt] = wmma_bf16(pa, vb, of[dt]);
            }
        }
        __syncthreads();   // before next kt overwrites Ks/Vs
    }

    // normalize and write [b*T+qi][h*HD + d]
    #pragma unroll
    for (int dt = 0; dt < 4; ++dt) {
        const int dcol = dt * 16 + ln;
        #pragma unroll
        for (int r = 0; r < 8; ++r) {
            const int qi = qt * 64 + wv * 16 + r + hf * 8;
            out[(size_t)(b * T_ + qi) * D_ + hh * HD_ + dcol] =
                f2bf(of[dt][r] / lrow[r]);
        }
    }
}

// ---------------------------------------------------------------------------
// Host-side launcher
// ---------------------------------------------------------------------------
extern "C" void kernel_launch(void* const* d_in, const int* in_sizes, int n_in,
                              void* d_out, int out_size, void* d_ws, size_t ws_size,
                              hipStream_t stream) {
    (void)in_sizes; (void)n_in; (void)out_size; (void)ws_size;

    const int*   x      = (const int*)  d_in[0];
    const float* emb    = (const float*)d_in[1];
    const float* pe     = (const float*)d_in[2];
    const float* ln1_w  = (const float*)d_in[3];
    const float* ln1_b  = (const float*)d_in[4];
    const float* qkv_w  = (const float*)d_in[5];
    const float* proj_w = (const float*)d_in[6];
    const float* proj_b = (const float*)d_in[7];
    const float* ln2_w  = (const float*)d_in[8];
    const float* ln2_b  = (const float*)d_in[9];
    const float* fc1_w  = (const float*)d_in[10];
    const float* fc1_b  = (const float*)d_in[11];
    const float* fc2_w  = (const float*)d_in[12];
    const float* fc2_b  = (const float*)d_in[13];
    const float* lnf_w  = (const float*)d_in[14];
    const float* lnf_b  = (const float*)d_in[15];
    float* logits = (float*)d_out;

    // workspace carve-up (all 16B aligned)
    char* ws = (char*)d_ws;
    float*  h      = (float*) ws;                             // 2048*1024 f32
    bf16_t* ln_out = (bf16_t*)(ws + 8388608);                 // 2048*1024 bf16
    bf16_t* qkvb   = (bf16_t*)(ws + 8388608 + 4194304);       // 2048*3072 bf16
    bf16_t* attnb  = (bf16_t*)(ws + 8388608 + 4194304 + 12582912);          // 2048*1024
    bf16_t* ffnb   = (bf16_t*)(ws + 8388608 + 4194304 + 12582912 + 4194304);// 2048*4096

    const dim3 blk256(256), blk128(128);

    embed_kernel<<<dim3(M_), blk256, 0, stream>>>(x, emb, pe, h);

    for (int l = 0; l < L_; ++l) {
        const float* Wqkv = qkv_w  + (size_t)l * 3 * D_ * D_;
        const float* Wp   = proj_w + (size_t)l * D_ * D_;
        const float* bp   = proj_b + (size_t)l * D_;
        const float* W1   = fc1_w  + (size_t)l * F_ * D_;
        const float* b1   = fc1_b  + (size_t)l * F_;
        const float* W2   = fc2_w  + (size_t)l * D_ * F_;
        const float* b2   = fc2_b  + (size_t)l * D_;

        layernorm_kernel<<<dim3(M_), blk256, 0, stream>>>(
            h, ln1_w + (size_t)l * D_, ln1_b + (size_t)l * D_, ln_out);

        gemm_bf16_kernel<0, false><<<dim3(3 * D_ / BN, M_ / BM), blk256, 0, stream>>>(
            ln_out, Wqkv, nullptr, M_, 3 * D_, D_, nullptr, qkvb, nullptr);

        flash_attn_kernel<<<dim3(T_ / 64, B_ * H_), blk128, 0, stream>>>(qkvb, attnb);

        gemm_bf16_kernel<2, true><<<dim3(D_ / BN, M_ / BM), blk256, 0, stream>>>(
            attnb, Wp, bp, M_, D_, D_, h, nullptr, h);

        layernorm_kernel<<<dim3(M_), blk256, 0, stream>>>(
            h, ln2_w + (size_t)l * D_, ln2_b + (size_t)l * D_, ln_out);

        gemm_bf16_kernel<1, true><<<dim3(F_ / BN, M_ / BM), blk256, 0, stream>>>(
            ln_out, W1, b1, M_, F_, D_, nullptr, ffnb, nullptr);

        gemm_bf16_kernel<2, true><<<dim3(D_ / BN, M_ / BM), blk256, 0, stream>>>(
            ffnb, W2, b2, M_, D_, F_, h, nullptr, h);
    }

    layernorm_kernel<<<dim3(M_), blk256, 0, stream>>>(h, lnf_w, lnf_b, ln_out);

    // tied-head logits: [2048, 32000] = ln_out @ emb^T
    gemm_bf16_kernel<3, false><<<dim3(V_ / BN, M_ / BM), blk256, 0, stream>>>(
        ln_out, emb, nullptr, M_, V_, D_, nullptr, nullptr, logits);
}
